// TransformerLayer_31756988186782
// MI455X (gfx1250) — matrix operand order repaired and checked by reading, not compiled, not run on hardware
//
#include <hip/hip_runtime.h>
#include <stdint.h>
#include <stddef.h>

// ---------------------------------------------------------------------------
// Transformer encoder layer for MI455X (gfx1250, wave32, WMMA).
//   B=2, S=2048, D=768, H=12, dk=64, D_FF=3072, fp32 in/out.
// Strategy: f16 data / f32 accumulate via v_wmma_f32_16x16x32_f16.
// GEMMs: per-wave register tiles (MT*16 x 64), weights pre-transposed [N][K]
// so B-fragments are contiguous; weights (~14 MB f16) stay resident in the
// 192 MB L2. Attention: single-pass flash attention, one wave per
// (b, h, 32-query tile), online softmax with half-wave shuffles, P tile
// re-laid out via a 2 KB LDS round trip.
// ---------------------------------------------------------------------------

#define D_MODEL 768
#define N_HEADS 12
#define D_FF    3072
#define SEQ     2048
#define BATCH   2
#define ROWS    (BATCH * SEQ)   // 4096
#define DK      64

typedef _Float16 h16;
typedef _Float16 v16h __attribute__((ext_vector_type(16)));
typedef float    v8f  __attribute__((ext_vector_type(8)));

struct alignas(16) U128 { unsigned int x[4]; };
struct alignas(32) U256 { U128 lo, hi; };

// Build a 16-element f16 WMMA fragment from two 16-byte chunks.
static __device__ __forceinline__ v16h load_frag(const void* p0, const void* p1) {
    U256 r;
    r.lo = *(const U128*)p0;
    r.hi = *(const U128*)p1;
    return __builtin_bit_cast(v16h, r);
}

static __device__ __forceinline__ v8f wmma_f16(v16h a, v16h b, v8f c) {
    return __builtin_amdgcn_wmma_f32_16x16x32_f16(false, a, false, b, (short)0, c,
                                                  false, false);
}

static __device__ __forceinline__ float gelu_exact(float x) {
    return 0.5f * x * (1.0f + erff(x * 0.70710678118654752f));
}

// ---------------------------------------------------------------------------
// Elementwise f32 -> f16
// ---------------------------------------------------------------------------
__global__ __launch_bounds__(256)
void cvt_f16_kernel(const float* __restrict__ in, h16* __restrict__ out, int n) {
    int i = blockIdx.x * 256 + threadIdx.x;
    if (i < n) out[i] = (h16)in[i];
}

// in[K][N] f32 (row-major)  ->  out[N][K] f16   (B-operand friendly layout)
__global__ __launch_bounds__(256)
void transpose_cvt_kernel(const float* __restrict__ in, h16* __restrict__ out,
                          int K, int N) {
    int i = blockIdx.x * 256 + threadIdx.x;
    if (i >= K * N) return;
    int kk = i / N;
    int n  = i % N;
    out[(size_t)n * K + kk] = (h16)in[i];
}

// ---------------------------------------------------------------------------
// GEMM: C[M,N] = act(A[M,K]_f16 @ W^T  + bias), W stored as WT[N][K]_f16.
// One wave per (MT*16) x 64 output tile: MT (M) x 4 (N) accumulators.
// MT=4 -> 16 WMMA per 16 b128 loads (~32 FLOP per L2 byte).
// ---------------------------------------------------------------------------
template <int MT>
__global__ __launch_bounds__(32)
void gemm_f16_kernel(const h16* __restrict__ A, const h16* __restrict__ WT,
                     const float* __restrict__ bias,
                     float* __restrict__ Cf, h16* __restrict__ Ch,
                     int M, int N, int K, int act_gelu) {
    const int lane = threadIdx.x & 31;
    const int half = lane >> 4;     // 0: lanes 0-15, 1: lanes 16-31
    const int lrow = lane & 15;
    const int cbase = blockIdx.x * 64;
    const int rbase = blockIdx.y * (MT * 16);

    const v8f zero = {0.f, 0.f, 0.f, 0.f, 0.f, 0.f, 0.f, 0.f};
    v8f acc[MT][4];
#pragma unroll
    for (int m = 0; m < MT; ++m)
#pragma unroll
        for (int n = 0; n < 4; ++n) acc[m][n] = zero;

    for (int k = 0; k < K; k += 32) {
        v16h af[MT];
#pragma unroll
        for (int m = 0; m < MT; ++m) {
            const h16* base = A + (size_t)(rbase + m * 16 + lrow) * K + k + half * 8;
            af[m] = load_frag(base, base + 16);
        }
#pragma unroll
        for (int n = 0; n < 4; ++n) {
            const h16* base = WT + (size_t)(cbase + n * 16 + lrow) * K + k + half * 8;
            v16h bf = load_frag(base, base + 16);
#pragma unroll
            for (int m = 0; m < MT; ++m) acc[m][n] = wmma_f16(af[m], bf, acc[m][n]);
        }
    }

#pragma unroll
    for (int n = 0; n < 4; ++n) {
        const int col = cbase + n * 16 + lrow;
        const float bv = bias[col];
#pragma unroll
        for (int m = 0; m < MT; ++m) {
#pragma unroll
            for (int i = 0; i < 8; ++i) {
                const int row = rbase + m * 16 + i + half * 8;
                float v = acc[m][n][i] + bv;
                if (act_gelu) v = gelu_exact(v);
                const size_t idx = (size_t)row * N + col;
                if (Cf) Cf[idx] = v;
                if (Ch) Ch[idx] = (h16)v;
            }
        }
    }
}

// ---------------------------------------------------------------------------
// Split fused QKV [ROWS][3*D] into Q,K: [B,H,S,dk] and V^T: [B,H,dk,S] (f16).
// ---------------------------------------------------------------------------
__global__ __launch_bounds__(256)
void split_heads_kernel(const h16* __restrict__ qkv,
                        h16* __restrict__ Q, h16* __restrict__ Km,
                        h16* __restrict__ VT) {
    int i = blockIdx.x * 256 + threadIdx.x;
    if (i >= ROWS * D_MODEL) return;
    const int row = i / D_MODEL;
    const int c   = i % D_MODEL;
    const int b = row / SEQ, s = row % SEQ;
    const int h = c / DK,   d = c % DK;
    const size_t head = ((size_t)(b * N_HEADS + h)) * SEQ * DK;
    const h16* src = qkv + (size_t)row * (3 * D_MODEL);
    Q [head + (size_t)s * DK + d] = src[c];
    Km[head + (size_t)s * DK + d] = src[D_MODEL + c];
    VT[head + (size_t)d * SEQ + s] = src[2 * D_MODEL + c];
}

// ---------------------------------------------------------------------------
// Flash attention: one wave per (b, h, 32-query tile).
// Q,K: [B,H,S,64], VT: [B,H,64,S].  O merged: [ROWS][D_MODEL] f16.
// ---------------------------------------------------------------------------
__global__ __launch_bounds__(32)
void attn_kernel(const h16* __restrict__ Q, const h16* __restrict__ Kmat,
                 const h16* __restrict__ VT, h16* __restrict__ O) {
    const int lane = threadIdx.x & 31;
    const int half = lane >> 4;
    const int lrow = lane & 15;
    const int qbase = blockIdx.x * 32;
    const int h = blockIdx.y, b = blockIdx.z;
    const size_t head = ((size_t)(b * N_HEADS + h)) * SEQ * DK;
    const h16* q  = Q    + head;
    const h16* kk = Kmat + head;
    const h16* vt = VT   + head;

    __shared__ __align__(16) h16 pshm[32 * 32];   // 2 KB P tile

    // Preload Q fragments: 2 M-tiles x 2 dk-chunks (dk=64 -> K=32 x 2).
    v16h qf[2][2];
#pragma unroll
    for (int m = 0; m < 2; ++m)
#pragma unroll
        for (int c = 0; c < 2; ++c) {
            const h16* base = q + (size_t)(qbase + m * 16 + lrow) * DK + c * 32 + half * 8;
            qf[m][c] = load_frag(base, base + 16);
        }

    const v8f zero = {0.f, 0.f, 0.f, 0.f, 0.f, 0.f, 0.f, 0.f};
    v8f acco[2][4];
    float mstate[2][8], lstate[2][8];
#pragma unroll
    for (int m = 0; m < 2; ++m) {
#pragma unroll
        for (int n = 0; n < 4; ++n) acco[m][n] = zero;
#pragma unroll
        for (int i = 0; i < 8; ++i) { mstate[m][i] = -1e30f; lstate[m][i] = 0.f; }
    }
    const float scale = 0.125f;   // 1/sqrt(64)

    for (int kb = 0; kb < SEQ; kb += 32) {
        // S = Q @ K^T  (32 queries x 32 keys)
        v8f accs[2][2];
#pragma unroll
        for (int m = 0; m < 2; ++m)
#pragma unroll
            for (int kt = 0; kt < 2; ++kt) accs[m][kt] = zero;
#pragma unroll
        for (int c = 0; c < 2; ++c) {
#pragma unroll
            for (int kt = 0; kt < 2; ++kt) {
                const h16* base =
                    kk + (size_t)(kb + kt * 16 + lrow) * DK + c * 32 + half * 8;
                v16h kf = load_frag(base, base + 16);
#pragma unroll
                for (int m = 0; m < 2; ++m)
                    accs[m][kt] = wmma_f16(qf[m][c], kf, accs[m][kt]);
            }
        }

        // Online softmax update (row = i + 8*half, spread over 16 lanes = cols).
#pragma unroll
        for (int m = 0; m < 2; ++m) {
#pragma unroll
            for (int i = 0; i < 8; ++i) {
                float s0 = accs[m][0][i] * scale;
                float s1 = accs[m][1][i] * scale;
                float rmax = fmaxf(s0, s1);
#pragma unroll
                for (int msk = 8; msk >= 1; msk >>= 1)
                    rmax = fmaxf(rmax, __shfl_xor(rmax, msk, 32));
                const float mold = mstate[m][i];
                const float mnew = fmaxf(mold, rmax);
                const float corr = __expf(mold - mnew);
                const float p0 = __expf(s0 - mnew);
                const float p1 = __expf(s1 - mnew);
                float rs = p0 + p1;
#pragma unroll
                for (int msk = 8; msk >= 1; msk >>= 1)
                    rs += __shfl_xor(rs, msk, 32);
                lstate[m][i] = lstate[m][i] * corr + rs;
                mstate[m][i] = mnew;
#pragma unroll
                for (int n = 0; n < 4; ++n) acco[m][n][i] *= corr;
                const int prow = m * 16 + i + half * 8;
                pshm[prow * 32 + lrow]      = (h16)p0;
                pshm[prow * 32 + 16 + lrow] = (h16)p1;
            }
        }
        __syncthreads();   // single wave: layout shuffle C-layout -> A-frag via LDS

        // O += P @ V   (K-dim = 32 keys)
        v16h pf[2];
#pragma unroll
        for (int m = 0; m < 2; ++m) {
            const h16* base = &pshm[(m * 16 + lrow) * 32 + half * 8];
            pf[m] = load_frag(base, base + 16);
        }
#pragma unroll
        for (int n = 0; n < 4; ++n) {
            const h16* vb = vt + (size_t)(n * 16 + lrow) * SEQ + kb + half * 8;
            v16h vf = load_frag(vb, vb + 16);
#pragma unroll
            for (int m = 0; m < 2; ++m) acco[m][n] = wmma_f16(pf[m], vf, acco[m][n]);
        }
        __syncthreads();
    }

    // Normalize and write merged heads: O[b*S+s][h*64 + d]
#pragma unroll
    for (int m = 0; m < 2; ++m) {
#pragma unroll
        for (int i = 0; i < 8; ++i) {
            const float inv = 1.0f / lstate[m][i];
            const int row = b * SEQ + qbase + m * 16 + i + half * 8;
#pragma unroll
            for (int n = 0; n < 4; ++n) {
                const int col = h * DK + n * 16 + lrow;
                O[(size_t)row * D_MODEL + col] = (h16)(acco[m][n][i] * inv);
            }
        }
    }
}

// ---------------------------------------------------------------------------
// out = LayerNorm(X + R) * g + be ; fp32 out, optional f16 copy.
// One 256-thread block per row of 768.
// ---------------------------------------------------------------------------
__global__ __launch_bounds__(256)
void add_ln_kernel(const float* __restrict__ X, const float* __restrict__ R,
                   const float* __restrict__ g, const float* __restrict__ be,
                   float* __restrict__ Of, h16* __restrict__ Oh) {
    __shared__ float red[8];
    const int row = blockIdx.x;
    const int t = threadIdx.x;
    const float* x = X + (size_t)row * D_MODEL;
    const float* r = R + (size_t)row * D_MODEL;

    float v[3];
    float s = 0.f;
#pragma unroll
    for (int j = 0; j < 3; ++j) {
        v[j] = x[t + j * 256] + r[t + j * 256];
        s += v[j];
    }
#pragma unroll
    for (int msk = 16; msk >= 1; msk >>= 1) s += __shfl_xor(s, msk, 32);
    if ((t & 31) == 0) red[t >> 5] = s;
    __syncthreads();
    float tot = 0.f;
#pragma unroll
    for (int i = 0; i < 8; ++i) tot += red[i];
    const float mu = tot * (1.0f / D_MODEL);
    __syncthreads();

    float sq = 0.f;
#pragma unroll
    for (int j = 0; j < 3; ++j) { float d = v[j] - mu; sq += d * d; }
#pragma unroll
    for (int msk = 16; msk >= 1; msk >>= 1) sq += __shfl_xor(sq, msk, 32);
    if ((t & 31) == 0) red[t >> 5] = sq;
    __syncthreads();
    tot = 0.f;
#pragma unroll
    for (int i = 0; i < 8; ++i) tot += red[i];
    const float rstd = rsqrtf(tot * (1.0f / D_MODEL) + 1e-5f);

#pragma unroll
    for (int j = 0; j < 3; ++j) {
        const int col = t + j * 256;
        const float o = (v[j] - mu) * rstd * g[col] + be[col];
        Of[(size_t)row * D_MODEL + col] = o;
        if (Oh) Oh[(size_t)row * D_MODEL + col] = (h16)o;
    }
}

// ---------------------------------------------------------------------------
// Host launch
// ---------------------------------------------------------------------------
extern "C" void kernel_launch(void* const* d_in, const int* in_sizes, int n_in,
                              void* d_out, int out_size, void* d_ws, size_t ws_size,
                              hipStream_t stream) {
    (void)in_sizes; (void)n_in; (void)out_size; (void)ws_size;

    const float* x     = (const float*)d_in[0];
    const float* w_qkv = (const float*)d_in[1];
    const float* b_qkv = (const float*)d_in[2];
    const float* w_out = (const float*)d_in[3];
    const float* b_out = (const float*)d_in[4];
    const float* w1    = (const float*)d_in[5];
    const float* b1    = (const float*)d_in[6];
    const float* w2    = (const float*)d_in[7];
    const float* b2    = (const float*)d_in[8];
    const float* g1    = (const float*)d_in[9];
    const float* be1   = (const float*)d_in[10];
    const float* g2    = (const float*)d_in[11];
    const float* be2   = (const float*)d_in[12];

    char* ws = (char*)d_ws;
    size_t off = 0;
    auto alloc = [&](size_t bytes) -> void* {
        void* p = ws + off;
        off += (bytes + 255) & ~(size_t)255;
        return p;
    };

    h16* xh    = (h16*)alloc((size_t)ROWS * D_MODEL * 2);      // later reused as y1_f16
    h16* wqkvT = (h16*)alloc((size_t)3 * D_MODEL * D_MODEL * 2);
    h16* woutT = (h16*)alloc((size_t)D_MODEL * D_MODEL * 2);
    h16* w1T   = (h16*)alloc((size_t)D_MODEL * D_FF * 2);
    h16* w2T   = (h16*)alloc((size_t)D_FF * D_MODEL * 2);
    h16* big   = (h16*)alloc((size_t)ROWS * D_FF * 2);         // qkv(18.9MB) then hidden(25.2MB)
    h16* qkvh  = big;
    h16* hidh  = big;
    h16* qh    = (h16*)alloc((size_t)ROWS * DK * N_HEADS * 2);
    h16* kh    = (h16*)alloc((size_t)ROWS * DK * N_HEADS * 2);
    h16* vTh   = (h16*)alloc((size_t)ROWS * DK * N_HEADS * 2);
    h16* oh    = (h16*)alloc((size_t)ROWS * D_MODEL * 2);
    float* t0  = (float*)alloc((size_t)ROWS * D_MODEL * 4);    // attn_out, then ffn_out
    float* y1  = (float*)alloc((size_t)ROWS * D_MODEL * 4);
    h16* y1h   = xh;                                           // alias (xh dead after QKV GEMM)

    const int TB = 256;
    // 1) Convert activations + transpose weights to f16.
    cvt_f16_kernel<<<(ROWS * D_MODEL + TB - 1) / TB, TB, 0, stream>>>(x, xh, ROWS * D_MODEL);
    transpose_cvt_kernel<<<(D_MODEL * 3 * D_MODEL + TB - 1) / TB, TB, 0, stream>>>(w_qkv, wqkvT, D_MODEL, 3 * D_MODEL);
    transpose_cvt_kernel<<<(D_MODEL * D_MODEL + TB - 1) / TB, TB, 0, stream>>>(w_out, woutT, D_MODEL, D_MODEL);
    transpose_cvt_kernel<<<(D_MODEL * D_FF + TB - 1) / TB, TB, 0, stream>>>(w1, w1T, D_MODEL, D_FF);
    transpose_cvt_kernel<<<(D_FF * D_MODEL + TB - 1) / TB, TB, 0, stream>>>(w2, w2T, D_FF, D_MODEL);

    // 2) QKV projection: [4096,768] @ [768,2304]  (big-N: 64x64 wave tiles)
    gemm_f16_kernel<4><<<dim3(3 * D_MODEL / 64, ROWS / 64), 32, 0, stream>>>(
        xh, wqkvT, b_qkv, nullptr, qkvh, ROWS, 3 * D_MODEL, D_MODEL, 0);

    // 3) Head split (V transposed for PV WMMA B-frags).
    split_heads_kernel<<<(ROWS * D_MODEL + TB - 1) / TB, TB, 0, stream>>>(qkvh, qh, kh, vTh);

    // 4) Flash attention, merged-head output.
    attn_kernel<<<dim3(SEQ / 32, N_HEADS, BATCH), 32, 0, stream>>>(qh, kh, vTh, oh);

    // 5) Output projection (small-N: 32x64 wave tiles for occupancy).
    gemm_f16_kernel<2><<<dim3(D_MODEL / 64, ROWS / 32), 32, 0, stream>>>(
        oh, woutT, b_out, t0, nullptr, ROWS, D_MODEL, D_MODEL, 0);

    // 6) y1 = LN(x + attn_out)
    add_ln_kernel<<<ROWS, 256, 0, stream>>>(x, t0, g1, be1, y1, y1h);

    // 7) FFN up + exact GELU (big-N: 64x64 wave tiles).
    gemm_f16_kernel<4><<<dim3(D_FF / 64, ROWS / 64), 32, 0, stream>>>(
        y1h, w1T, b1, nullptr, hidh, ROWS, D_FF, D_MODEL, 1);

    // 8) FFN down (small-N: 32x64 wave tiles).
    gemm_f16_kernel<2><<<dim3(D_MODEL / 64, ROWS / 32), 32, 0, stream>>>(
        hidh, w2T, b2, t0, nullptr, ROWS, D_MODEL, D_FF, 0);

    // 9) out = LN(y1 + ffn_out)
    add_ln_kernel<<<ROWS, 256, 0, stream>>>(y1, t0, g2, be2, (float*)d_out, nullptr);
}
